// DorefaConv2d_36644660969538
// MI455X (gfx1250) — 8-bit matrix operand order repaired and checked by reading, compiled, not run on hardware
//
#include <hip/hip_runtime.h>

typedef __attribute__((ext_vector_type(8))) int v8i;

#define NN 8
#define HH 512
#define WW 512
#define CIN 16
#define COUT 32
#define NPIX (NN*HH*WW)
#define NACT (NPIX*CIN)
#define NWELT (3*3*CIN*COUT)
#define BPACK_DWORDS (3*2*32*8)

__device__ __forceinline__ unsigned int quant4(float4 v) {
    unsigned int q0 = (unsigned int)(int)rintf(fminf(fabsf(v.x), 1.0f) * 7.0f);
    unsigned int q1 = (unsigned int)(int)rintf(fminf(fabsf(v.y), 1.0f) * 7.0f);
    unsigned int q2 = (unsigned int)(int)rintf(fminf(fabsf(v.z), 1.0f) * 7.0f);
    unsigned int q3 = (unsigned int)(int)rintf(fminf(fabsf(v.w), 1.0f) * 7.0f);
    return q0 | (q1 << 8) | (q2 << 16) | (q3 << 24);
}

__global__ void __launch_bounds__(256) quant_act_kernel(const float4* __restrict__ x,
                                                        uint4* __restrict__ aq) {
    const int i = blockIdx.x * blockDim.x + threadIdx.x;
    const float4 v0 = x[4 * i + 0];
    const float4 v1 = x[4 * i + 1];
    const float4 v2 = x[4 * i + 2];
    const float4 v3 = x[4 * i + 3];
    uint4 q;
    q.x = quant4(v0); q.y = quant4(v1); q.z = quant4(v2); q.w = quant4(v3);
    aq[i] = q;
}

__global__ void __launch_bounds__(256) prep_kernel(const float* __restrict__ Wt,
                                                   unsigned int* __restrict__ bpack,
                                                   float* __restrict__ e7) {
    __shared__ float red[256];
    const int tid = threadIdx.x;
    float s = 0.0f;
    for (int i = tid; i < NWELT; i += 256) s += fabsf(Wt[i]);
    red[tid] = s;
    __syncthreads();
    for (int off = 128; off > 0; off >>= 1) {
        if (tid < off) red[tid] += red[tid + off];
        __syncthreads();
    }
    if (tid == 0) *e7 = (red[0] / (float)NWELT) * (1.0f / 7.0f);

    for (int idx = tid; idx < BPACK_DWORDS; idx += 256) {
        const int v  = idx & 7;
        const int l  = (idx >> 3) & 31;
        const int nt = (idx >> 8) & 1;
        const int kc = idx >> 9;
        const int o  = nt * 16 + (l & 15);
        unsigned int dw = 0;
        for (int byte = 0; byte < 4; ++byte) {
            const int Klocal = (v >> 1) * 16 + ((l >= 16) ? 8 : 0) + (v & 1) * 4 + byte;
            const int K = kc * 64 + Klocal;
            int sgn = 0;
            if (K < 144) {
                const float w = Wt[K * COUT + o];
                sgn = (w > 0.0f) ? 1 : ((w < 0.0f) ? -1 : 0);
            }
            dw |= ((unsigned int)(sgn & 0xFF)) << (8 * byte);
        }
        bpack[idx] = dw;
    }
}

__global__ void __launch_bounds__(256) conv_wmma_kernel(const unsigned char* __restrict__ aq,
                                                        const unsigned int* __restrict__ bpack,
                                                        const float* __restrict__ e7p,
                                                        const float* __restrict__ bias,
                                                        float* __restrict__ out) {
    const int lane = threadIdx.x & 31;
    const int tile = __builtin_amdgcn_readfirstlane(blockIdx.x * 8 + (threadIdx.x >> 5));
    const int p0 = tile << 4;
    const int w0 = p0 & (WW - 1);
    const int h  = (p0 >> 9) & (HH - 1);
    const int n  = p0 >> 18;

    const int m = lane & 15;
    const int chalf = (lane >> 4) << 3;

    v8i a0, a1, a2;
    const bool interior = (h >= 1) && (h <= HH - 2) && (w0 >= 16) && (w0 <= WW - 32);

    if (interior) {
        const unsigned char* base = aq + (((n * HH + h) * WW + w0 + m) * CIN + chalf);
#pragma unroll
        for (int kc = 0; kc < 3; ++kc) {
            v8i a;
#pragma unroll
            for (int j = 0; j < 4; ++j) {
                const int t = kc * 4 + j;
                uint2 d; d.x = 0; d.y = 0;
                if (t < 9) {
                    const int off = ((t / 3 - 1) * WW + (t % 3 - 1)) * CIN;
                    d = *(const uint2*)(base + off);
                }
                a[2 * j]     = (int)d.x;
                a[2 * j + 1] = (int)d.y;
            }
            if (kc == 0) a0 = a; else if (kc == 1) a1 = a; else a2 = a;
        }
    } else {
        const int xw = w0 + m;
#pragma unroll
        for (int kc = 0; kc < 3; ++kc) {
            v8i a;
#pragma unroll
            for (int j = 0; j < 4; ++j) {
                const int t = kc * 4 + j;
                unsigned int lo = 0, hi = 0;
                if (t < 9) {
                    const int y = h + (t / 3) - 1;
                    const int x = xw + (t % 3) - 1;
                    if ((unsigned)y < (unsigned)HH && (unsigned)x < (unsigned)WW) {
                        const uint2 d = *(const uint2*)(aq + (((n * HH + y) * WW + x) * CIN + chalf));
                        lo = d.x; hi = d.y;
                    }
                }
                a[2 * j]     = (int)lo;
                a[2 * j + 1] = (int)hi;
            }
            if (kc == 0) a0 = a; else if (kc == 1) a1 = a; else a2 = a;
        }
    }

    v8i acc0 = {0, 0, 0, 0, 0, 0, 0, 0};
    v8i acc1 = {0, 0, 0, 0, 0, 0, 0, 0};

#pragma unroll
    for (int kc = 0; kc < 3; ++kc) {
        const v8i a = (kc == 0) ? a0 : (kc == 1) ? a1 : a2;
        const uint4* bp0 = (const uint4*)(bpack + ((kc * 2 + 0) * 32 + lane) * 8);
        const uint4* bp1 = (const uint4*)(bpack + ((kc * 2 + 1) * 32 + lane) * 8);
        const uint4 c0 = bp0[0], c1 = bp0[1];
        const uint4 d0 = bp1[0], d1 = bp1[1];
        v8i b0, b1;
        b0[0] = (int)c0.x; b0[1] = (int)c0.y; b0[2] = (int)c0.z; b0[3] = (int)c0.w;
        b0[4] = (int)c1.x; b0[5] = (int)c1.y; b0[6] = (int)c1.z; b0[7] = (int)c1.w;
        b1[0] = (int)d0.x; b1[1] = (int)d0.y; b1[2] = (int)d0.z; b1[3] = (int)d0.w;
        b1[4] = (int)d1.x; b1[5] = (int)d1.y; b1[6] = (int)d1.z; b1[7] = (int)d1.w;

        acc0 = __builtin_amdgcn_wmma_i32_16x16x64_iu8(false, a, true, b0, acc0, false, false);
        acc1 = __builtin_amdgcn_wmma_i32_16x16x64_iu8(false, a, true, b1, acc1, false, false);
    }

    const float scale = *e7p;
    const int o0 = lane & 15;
    const float bb0 = bias[o0];
    const float bb1 = bias[16 + o0];
    const int mbase = (lane >> 4) << 3;
#pragma unroll
    for (int k = 0; k < 8; ++k) {
        const int p = p0 + mbase + k;
        out[p * COUT + o0]      = (float)acc0[k] * scale + bb0;
        out[p * COUT + 16 + o0] = (float)acc1[k] * scale + bb1;
    }
}

extern "C" void kernel_launch(void* const* d_in, const int* in_sizes, int n_in,
                              void* d_out, int out_size, void* d_ws, size_t ws_size,
                              hipStream_t stream) {
    const float* x  = (const float*)d_in[0];
    const float* Wt = (const float*)d_in[1];
    const float* b  = (const float*)d_in[2];
    float* out = (float*)d_out;

    unsigned char* aq    = (unsigned char*)d_ws;
    unsigned int*  bpack = (unsigned int*)((char*)d_ws + NACT);
    float*         e7    = (float*)((char*)d_ws + NACT + BPACK_DWORDS * 4);

    quant_act_kernel<<<NACT / 16 / 256, 256, 0, stream>>>((const float4*)x, (uint4*)aq);
    prep_kernel<<<1, 256, 0, stream>>>(Wt, bpack, e7);
    conv_wmma_kernel<<<NPIX / 16 / 8, 256, 0, stream>>>(aq, bpack, e7, b, out);
}
